// StochasticRNNPB_76716705841383
// MI455X (gfx1250) — matrix operand order repaired and checked by reading, not compiled, not run on hardware
//
#include <hip/hip_runtime.h>

// ---------------------------------------------------------------------------
// StochasticRNNPB: 2-layer LSTM, H=1024, B=64, D=64, PB=16, T=256, wave32 WMMA
// Persistent-kernel design for MI455X (gfx1250):
//   * weights bf16-packed once per call into WMMA B-frag order, streamed with
//     non-temporal global_load_b128 (L2-resident, no WGP$ pollution)
//   * activations staged workgroup-wide into LDS via CDNA5 async-to-LDS
//     (global_load_async_to_lds_b128 + s_wait_asynccnt), double-buffered in
//     16KB chunks; WMMA consumes via ds_load_b128 (one copy serves 8 waves)
//   * 64 workgroups x 256 threads (8 waves); block b owns hidden cols
//     [16b,16b+16) of both layers; wave = (gate, M-half)
//   * gate bias folded into WMMA accumulator init
//   * h ping-pongs through packed bf16 A-frag buffers in global (L2-hot);
//     cell state c lives in LDS for the whole run
//   * software grid barrier (atomic + s_sleep), 3 per timestep
// ---------------------------------------------------------------------------

typedef __attribute__((ext_vector_type(16))) __bf16 v16bf;
typedef __attribute__((ext_vector_type(8)))  float  v8f;

#define T_STEPS 256
#define Bsz     64
#define Hsz     1024
#define Dsz     64
#define PBsz    16
#define NGATES  4096           // 4*H
#define NT_H    256            // 4096/16 gate column tiles
#define KBH     32             // 1024/32 k-blocks for H-wide operands
#define KB0     3              // D+PB=80 padded to 3 k-blocks (96)
#define NBLK    64             // persistent grid size
#define CH      4              // k-blocks per staged LDS chunk
#define AH_ELEMS (4*KBH*512)   // packed A frags: 4 mtiles * 32 kb * 32 lanes * 16
#define LDSA_ELEMS (4*CH*512)  // one staging buffer: 4 mtiles * CH kb * 512

// ---------------- packing: fp32 [N,K] row-major -> bf16 WMMA B fragments ----
// packed idx = ((ntile*KB + kb)*32 + lane)*16 + j
// element (k,n): n = ntile*16 + (lane&15); k = kb*32 + (j>>3)*16 + (lane>>4)*8 + (j&7)
__global__ void pack_b_kernel(const float* __restrict__ src,
                              __bf16* __restrict__ dst,
                              int ntiles, int KB, int Ksrc) {
    int total  = ntiles * KB * 512;
    int stride = gridDim.x * blockDim.x;
    for (int idx = blockIdx.x * blockDim.x + threadIdx.x; idx < total; idx += stride) {
        int j  = idx & 15;
        int L  = (idx >> 4) & 31;
        int kb = (idx >> 9) % KB;
        int nt = (idx >> 9) / KB;
        int k  = kb * 32 + ((j >> 3) << 4) + (L >> 4) * 8 + (j & 7);
        int n  = nt * 16 + (L & 15);
        float v = (k < Ksrc) ? src[(size_t)n * Ksrc + k] : 0.0f;
        dst[idx] = (__bf16)v;
    }
}

// ---------------- misc setup: biases, PB part of X0, tuple-tail outputs -----
__global__ void setup_misc_kernel(const int* __restrict__ label,
                                  const float* __restrict__ mu_pb,
                                  const float* __restrict__ logvar_pb,
                                  const float* __restrict__ bi0, const float* __restrict__ bh0,
                                  const float* __restrict__ bi1, const float* __restrict__ bh1,
                                  float* __restrict__ bsum0, float* __restrict__ bsum1,
                                  __bf16* __restrict__ A0p,
                                  float* __restrict__ tail) {
    int stride = gridDim.x * blockDim.x;
    int t0 = blockIdx.x * blockDim.x + threadIdx.x;
    for (int i = t0; i < NGATES; i += stride) {
        bsum0[i] = bi0[i] + bh0[i];
        bsum1[i] = bi1[i] + bh1[i];
    }
    // packed X0 fragments: y-part (k<64) zeroed (y0=0), pb part, pad zeros
    for (int idx = t0; idx < 4 * KB0 * 512; idx += stride) {
        int j  = idx & 15;
        int L  = (idx >> 4) & 31;
        int kb = (idx >> 9) % KB0;
        int mt = (idx >> 9) / KB0;
        int k  = kb * 32 + ((j >> 3) << 4) + (L >> 4) * 8 + (j & 7);
        int m  = mt * 16 + (L & 15);
        float v = 0.0f;
        if (k >= Dsz && k < Dsz + PBsz) v = mu_pb[label[m] * PBsz + (k - Dsz)];
        A0p[idx] = (__bf16)v;
    }
    // tuple tail: sequence_label, pb, mu_sel, logvar_sel
    for (int i = t0; i < Bsz; i += stride) tail[i] = (float)label[i];
    for (int i = t0; i < Bsz * PBsz; i += stride) {
        int b = i >> 4, p = i & 15;
        float mv = mu_pb[label[b] * PBsz + p];
        tail[Bsz + i]              = mv;   // pb
        tail[Bsz + Bsz*PBsz + i]   = mv;   // mu_sel
        tail[Bsz + 2*Bsz*PBsz + i] = logvar_pb[label[b] * PBsz + p];
    }
}

// ---------------- device helpers -------------------------------------------
__device__ __forceinline__ v8f wmma_bf16(v16bf a, v16bf b, v8f c) {
    return __builtin_amdgcn_wmma_f32_16x16x32_bf16(false, a, false, b, (short)0, c,
                                                   false, false);
}

// CDNA5 async global->LDS copy, 16B per lane (ISA 15.18.3 opcode 98), ASYNCcnt
__device__ __forceinline__ void async_copy_b128(unsigned lds_off, const void* gsrc) {
    asm volatile("global_load_async_to_lds_b128 %0, %1, off"
                 :: "v"(lds_off), "v"(gsrc) : "memory");
}
__device__ __forceinline__ void wait_async_le(int) {}
__device__ __forceinline__ void wait_async_4() {
    asm volatile("s_wait_asynccnt 0x4" ::: "memory");
}
__device__ __forceinline__ void wait_async_0() {
    asm volatile("s_wait_asynccnt 0x0" ::: "memory");
}

__device__ __forceinline__ void grid_sync(unsigned* cnt, unsigned tgt) {
    __threadfence();
    __syncthreads();
    if (threadIdx.x == 0) {
        atomicAdd(cnt, 1u);
        while (*(volatile unsigned*)cnt < tgt) __builtin_amdgcn_s_sleep(2);
    }
    __syncthreads();
    __threadfence();
}

__device__ __forceinline__ float sigmoidf_(float x) {
    return 1.0f / (1.0f + __expf(-x));
}

// direct (non-staged) fragment GEMM for the small K=96 input stream
template <int KB>
__device__ __forceinline__ void gemm_frags(v8f& acc0, v8f& acc1,
                                           const v16bf* __restrict__ W,
                                           const v16bf* __restrict__ A0,
                                           const v16bf* __restrict__ A1) {
    v16bf b  = __builtin_nontemporal_load(W);
    v16bf a0 = *A0;
    v16bf a1 = *A1;
#pragma unroll
    for (int kb = 0; kb < KB; ++kb) {
        v16bf bn = {}, a0n = {}, a1n = {};
        if (kb + 1 < KB) {
            bn  = __builtin_nontemporal_load(W + (kb + 1) * 32);
            a0n = A0[(kb + 1) * 32];
            a1n = A1[(kb + 1) * 32];
        }
        acc0 = wmma_bf16(a0, b, acc0);
        acc1 = wmma_bf16(a1, b, acc1);
        b = bn; a0 = a0n; a1 = a1n;
    }
}

// staged fragment GEMM over K = KB*32: activations flow global->LDS via async
// copies (double-buffered CH-kb chunks, all 8 waves cooperate), weights stream
// non-temporally from global, WMMA reads A frags with ds_load_b128.
template <int KB>
__device__ __forceinline__ void gemm_frags_staged(v8f& acc0, v8f& acc1,
                                                  const __bf16* __restrict__ Wp, int nt,
                                                  const __bf16* __restrict__ Ap,
                                                  __bf16* lds_a,      // [2][LDSA_ELEMS]
                                                  int wave, int lane, int mt0) {
    static_assert(KB % CH == 0, "KB must be chunk-aligned");
    constexpr int NCHUNK = KB / CH;
    const v16bf* W = (const v16bf*)Wp + (size_t)nt * KB * 32 + lane;

    // per-chunk copy share: 32 B128 copies; wave w issues q = 4w..4w+3
    auto issue = [&](int c) {
#pragma unroll
        for (int i = 0; i < 4; ++i) {
            int q    = wave * 4 + i;
            int frag = q >> 1;                 // 0..15 = mt*CH + kc
            int half = q & 1;                  // which 16B of the 32B frag line
            int mt   = frag >> 2;
            int kc   = frag & (CH - 1);
            const char* g = (const char*)Ap +
                ((((size_t)mt * KB + (c * CH + kc)) * 32 + lane) * 32 + half * 16);
            unsigned l = (unsigned)(uintptr_t)(lds_a + (size_t)(c & 1) * LDSA_ELEMS
                                               + (((mt * CH + kc) * 32 + lane) * 16 + half * 8));
            async_copy_b128(l, g);
        }
    };

    issue(0);
    for (int c = 0; c < NCHUNK; ++c) {
        if (c + 1 < NCHUNK) issue(c + 1);      // keep next chunk in flight
        // preload this chunk's weight frags (overlaps the async wait)
        v16bf bq[CH];
#pragma unroll
        for (int kc = 0; kc < CH; ++kc)
            bq[kc] = __builtin_nontemporal_load(W + (c * CH + kc) * 32);
        if (c + 1 < NCHUNK) wait_async_4();    // my chunk-c copies done (<=4 left)
        else                wait_async_0();
        __syncthreads();                        // all waves' chunk-c copies visible
        const v16bf* ls = (const v16bf*)(lds_a + (size_t)(c & 1) * LDSA_ELEMS);
#pragma unroll
        for (int kc = 0; kc < CH; ++kc) {
            v16bf a0 = ls[((mt0)     * CH + kc) * 32 + lane];
            v16bf a1 = ls[((mt0 + 1) * CH + kc) * 32 + lane];
            acc0 = wmma_bf16(a0, bq[kc], acc0);
            acc1 = wmma_bf16(a1, bq[kc], acc1);
        }
        __syncthreads();                        // chunk-c buffer free for reuse
    }
}

// one LSTM layer for this block's 16 hidden columns
template <int KBX>
__device__ __forceinline__ void lstm_layer(const __bf16* __restrict__ Axp,
                                           const __bf16* __restrict__ Wxp,
                                           const __bf16* __restrict__ Ahp_prev,
                                           const __bf16* __restrict__ Whp,
                                           const float* __restrict__ bsum,
                                           float* lds_g, float* lds_c, __bf16* lds_a,
                                           __bf16* __restrict__ Ahp_out,
                                           int blk, int wave, int lane, int tid) {
    const int gate = wave >> 1;
    const int mt0  = (wave & 1) * 2;           // mtiles {0,1} or {2,3}
    const int nt   = gate * 64 + blk;          // gate column tile in [0,256)

    // bias folded into accumulator init: per-column (== per-lane) constant
    const float bv = bsum[gate * Hsz + blk * 16 + (lane & 15)];
    v8f acc0 = {bv, bv, bv, bv, bv, bv, bv, bv};
    v8f acc1 = acc0;

    if constexpr (KBX == KBH) {
        gemm_frags_staged<KBH>(acc0, acc1, Wxp, nt, Axp, lds_a, wave, lane, mt0);
    } else {
        gemm_frags<KBX>(acc0, acc1,
                        (const v16bf*)Wxp + (size_t)nt * KBX * 32 + lane,
                        (const v16bf*)Axp + (size_t)mt0 * KBX * 32 + lane,
                        (const v16bf*)Axp + (size_t)(mt0 + 1) * KBX * 32 + lane);
    }
    gemm_frags_staged<KBH>(acc0, acc1, Whp, nt, Ahp_prev, lds_a, wave, lane, mt0);

    // spill gate tiles to LDS: D-frag element r -> m = mt*16 + (lane>>4)*8 + r
    const int n = lane & 15;
#pragma unroll
    for (int r = 0; r < 8; ++r) {
        int mA = mt0 * 16 + (lane >> 4) * 8 + r;
        int mB = (mt0 + 1) * 16 + (lane >> 4) * 8 + r;
        lds_g[gate * 1024 + mA * 16 + n] = acc0[r];
        lds_g[gate * 1024 + mB * 16 + n] = acc1[r];
    }
    __syncthreads();

    // LSTM cell update: 1024 elements over 256 threads (bias already in gates)
#pragma unroll
    for (int r = 0; r < 4; ++r) {
        int e   = tid + r * 256;               // e = m*16 + ncol
        int m   = e >> 4;
        int col = blk * 16 + (e & 15);         // hidden index
        float gi = lds_g[0 * 1024 + e];
        float gf = lds_g[1 * 1024 + e];
        float gg = lds_g[2 * 1024 + e];
        float go = lds_g[3 * 1024 + e];
        float c  = sigmoidf_(gf) * lds_c[e] + sigmoidf_(gi) * tanhf(gg);
        lds_c[e] = c;
        float h  = sigmoidf_(go) * tanhf(c);
        // scatter h into packed bf16 A fragments (k = col)
        int kb  = col >> 5, kib = col & 31;
        int Lp  = ((kib >> 3) & 1) * 16 + (m & 15);
        int j   = ((kib >> 4) << 3) + (kib & 7);
        Ahp_out[(((size_t)(m >> 4) * KBH + kb) * 32 + Lp) * 16 + j] = (__bf16)h;
    }
    __syncthreads();
}

// ---------------- persistent LSTM kernel ------------------------------------
__global__ __launch_bounds__(256)
void lstm_persistent_kernel(const __bf16* __restrict__ Wih0p,
                            const __bf16* __restrict__ Whh0p,
                            const __bf16* __restrict__ Wih1p,
                            const __bf16* __restrict__ Whh1p,
                            const __bf16* __restrict__ Wlinp,
                            const float* __restrict__ bsum0,
                            const float* __restrict__ bsum1,
                            const float* __restrict__ b_lin,
                            __bf16* __restrict__ A0p,
                            __bf16* __restrict__ Ah0p,   // [2][AH_ELEMS]
                            __bf16* __restrict__ Ah1p,   // [2][AH_ELEMS]
                            float* __restrict__ out,     // [T][B][D]
                            unsigned* __restrict__ bar) {
    __shared__ float  lds_g[4 * 1024];                   // 16KB gate exchange
    __shared__ float  lds_c[2][1024];                    //  8KB cell state
    __shared__ __bf16 lds_a[2 * LDSA_ELEMS];             // 32KB async A staging

    const int tid  = threadIdx.x;
    const int wave = tid >> 5;
    const int lane = tid & 31;
    const int blk  = blockIdx.x;

    for (int i = tid; i < 1024; i += 256) { lds_c[0][i] = 0.f; lds_c[1][i] = 0.f; }
    __syncthreads();

    unsigned bar_target = 0;

    for (int t = 0; t < T_STEPS; ++t) {
        const int cur = t & 1, prv = cur ^ 1;
        __bf16* ah0_cur = Ah0p + (size_t)cur * AH_ELEMS;
        __bf16* ah0_prv = Ah0p + (size_t)prv * AH_ELEMS;
        __bf16* ah1_cur = Ah1p + (size_t)cur * AH_ELEMS;
        __bf16* ah1_prv = Ah1p + (size_t)prv * AH_ELEMS;

        // ---- layer 0: X0 = [y(t-1), pb] (packed A0p), h0(t-1) ----
        lstm_layer<KB0>(A0p, Wih0p, ah0_prv, Whh0p, bsum0,
                        lds_g, lds_c[0], lds_a, ah0_cur, blk, wave, lane, tid);
        bar_target += NBLK; grid_sync(bar, bar_target);

        // ---- layer 1: X1 = h0(t), h1(t-1) ----
        lstm_layer<KBH>(ah0_cur, Wih1p, ah1_prv, Whh1p, bsum1,
                        lds_g, lds_c[1], lds_a, ah1_cur, blk, wave, lane, tid);
        bar_target += NBLK; grid_sync(bar, bar_target);

        // ---- output projection: y(t) = h1(t) @ Wlin^T + b_lin (block 0) ----
        if (blk == 0 && wave < 4) {
            const int wt = wave;                 // column tile of y, D=64 -> 4 tiles
            const float bl = b_lin[wt * 16 + (lane & 15)];
            v8f acc[4];
#pragma unroll
            for (int mt = 0; mt < 4; ++mt) acc[mt] = (v8f){bl, bl, bl, bl, bl, bl, bl, bl};
            const v16bf* Wl = (const v16bf*)Wlinp + (size_t)wt * KBH * 32 + lane;
            const v16bf* Ah = (const v16bf*)ah1_cur + lane;
#pragma unroll 4
            for (int kb = 0; kb < KBH; ++kb) {
                v16bf b = __builtin_nontemporal_load(Wl + kb * 32);
#pragma unroll
                for (int mt = 0; mt < 4; ++mt) {
                    v16bf a = Ah[((size_t)mt * KBH + kb) * 32];
                    acc[mt] = wmma_bf16(a, b, acc[mt]);
                }
            }
            const int n = wt * 16 + (lane & 15);
#pragma unroll
            for (int mt = 0; mt < 4; ++mt) {
#pragma unroll
                for (int r = 0; r < 8; ++r) {
                    int m = mt * 16 + (lane >> 4) * 8 + r;
                    float y = acc[mt][r];
                    out[((size_t)t * Bsz + m) * Dsz + n] = y;
                    // feed y into next step's packed X0 (k = n in [0,64))
                    int kb  = n >> 5, kib = n & 31;
                    int Lp  = ((kib >> 3) & 1) * 16 + (m & 15);
                    int j   = ((kib >> 4) << 3) + (kib & 7);
                    A0p[(((size_t)(m >> 4) * KB0 + kb) * 32 + Lp) * 16 + j] = (__bf16)y;
                }
            }
        }
        bar_target += NBLK; grid_sync(bar, bar_target);
    }
}

// ---------------- host launch ------------------------------------------------
extern "C" void kernel_launch(void* const* d_in, const int* in_sizes, int n_in,
                              void* d_out, int out_size, void* d_ws, size_t ws_size,
                              hipStream_t stream) {
    (void)in_sizes; (void)n_in; (void)out_size; (void)ws_size;
    const int*   label   = (const int*)  d_in[0];
    const float* mu_pb   = (const float*)d_in[2];
    const float* logvar  = (const float*)d_in[3];
    const float* W_ih0   = (const float*)d_in[4];
    const float* W_hh0   = (const float*)d_in[5];
    const float* b_ih0   = (const float*)d_in[6];
    const float* b_hh0   = (const float*)d_in[7];
    const float* W_ih1   = (const float*)d_in[8];
    const float* W_hh1   = (const float*)d_in[9];
    const float* b_ih1   = (const float*)d_in[10];
    const float* b_hh1   = (const float*)d_in[11];
    const float* W_lin   = (const float*)d_in[12];
    const float* b_lin   = (const float*)d_in[13];
    float* out = (float*)d_out;

    // workspace carve-up (all offsets 256B aligned)
    char* ws = (char*)d_ws;
    size_t off = 0;
    auto take = [&](size_t bytes) {
        char* p = ws + off;
        off = (off + bytes + 255) & ~(size_t)255;
        return p;
    };
    unsigned* bar  = (unsigned*)take(256);
    float* bsum0   = (float*)take(NGATES * 4);
    float* bsum1   = (float*)take(NGATES * 4);
    __bf16* Wih0p  = (__bf16*)take((size_t)NT_H * KB0 * 512 * 2);
    __bf16* Whh0p  = (__bf16*)take((size_t)NT_H * KBH * 512 * 2);
    __bf16* Wih1p  = (__bf16*)take((size_t)NT_H * KBH * 512 * 2);
    __bf16* Whh1p  = (__bf16*)take((size_t)NT_H * KBH * 512 * 2);
    __bf16* Wlinp  = (__bf16*)take((size_t)4 * KBH * 512 * 2);
    __bf16* A0p    = (__bf16*)take((size_t)4 * KB0 * 512 * 2);
    __bf16* Ah0p   = (__bf16*)take((size_t)2 * AH_ELEMS * 2);
    __bf16* Ah1p   = (__bf16*)take((size_t)2 * AH_ELEMS * 2);

    hipMemsetAsync(bar, 0, 256, stream);
    hipMemsetAsync(Ah0p, 0, (size_t)2 * AH_ELEMS * 2, stream);
    hipMemsetAsync(Ah1p, 0, (size_t)2 * AH_ELEMS * 2, stream);

    pack_b_kernel<<<1024, 256, 0, stream>>>(W_ih0, Wih0p, NT_H, KB0, Dsz + PBsz);
    pack_b_kernel<<<4096, 256, 0, stream>>>(W_hh0, Whh0p, NT_H, KBH, Hsz);
    pack_b_kernel<<<4096, 256, 0, stream>>>(W_ih1, Wih1p, NT_H, KBH, Hsz);
    pack_b_kernel<<<4096, 256, 0, stream>>>(W_hh1, Whh1p, NT_H, KBH, Hsz);
    pack_b_kernel<<<256,  256, 0, stream>>>(W_lin, Wlinp, 4,    KBH, Hsz);

    setup_misc_kernel<<<64, 256, 0, stream>>>(label, mu_pb, logvar,
                                              b_ih0, b_hh0, b_ih1, b_hh1,
                                              bsum0, bsum1, A0p,
                                              out + (size_t)T_STEPS * Bsz * Dsz);

    lstm_persistent_kernel<<<NBLK, 256, 0, stream>>>(Wih0p, Whh0p, Wih1p, Whh1p, Wlinp,
                                                     bsum0, bsum1, b_lin,
                                                     A0p, Ah0p, Ah1p, out, bar);
}
